// MultiHeadAttention_84464826843578
// MI455X (gfx1250) — compile-verified, matrix operand order checked by reading, not run on hardware
//
#include <hip/hip_runtime.h>

// ---------- types ----------
typedef __attribute__((ext_vector_type(16))) __bf16 v16bf;
typedef __attribute__((ext_vector_type(8)))  float  v8f;
typedef __attribute__((ext_vector_type(4)))  unsigned int v4u;
typedef __attribute__((ext_vector_type(4)))  int v4i;

union FragU { v16bf v; v4u u[2]; };

__device__ inline unsigned short f2bf(float f) {
    unsigned u = __float_as_uint(f);
    u += 0x7FFFu + ((u >> 16) & 1u);   // round-to-nearest-even
    return (unsigned short)(u >> 16);
}

// ---------- CDNA5 async global->LDS copy (ASYNCcnt path), guarded ----------
#if defined(__AMDGCN__) && __has_builtin(__builtin_amdgcn_global_load_async_to_lds_b128)
#define HAVE_ASYNC 1
#else
#define HAVE_ASYNC 0
#endif

#define GLOBAL_AS __attribute__((address_space(1)))
#define LDS_AS    __attribute__((address_space(3)))

__device__ inline void cp_async_b128(const void* g, void* l) {
#if HAVE_ASYNC
    __builtin_amdgcn_global_load_async_to_lds_b128(
        (GLOBAL_AS v4i*)g, (LDS_AS v4i*)l, /*offset=*/0, /*cpol=*/0);
#else
    *(v4u*)l = *(const v4u*)g;
#endif
}

__device__ inline void cp_async_wait0() {
#if HAVE_ASYNC
#if __has_builtin(__builtin_amdgcn_s_wait_asynccnt)
    __builtin_amdgcn_s_wait_asynccnt(0);
#else
    asm volatile("s_wait_asynccnt 0" ::: "memory");
#endif
#endif
}

// Load a 16x32 bf16 A/B fragment (mirrored layouts) from row-major storage.
// Per CDNA5 ISA 7.12.2: lanes 0-15 hold K{k0..k0+7, k0+16..k0+23} of row,
// lanes 16-31 hold K{k0+8..k0+15, k0+24..k0+31}. Caller passes row incl. lane&15.
__device__ inline v16bf load_frag(const unsigned short* base, int stride, int row, int k0) {
    int lane  = threadIdx.x & 31;
    int khalf = (lane & 16) ? 8 : 0;
    const unsigned short* p = base + row * stride + k0 + khalf;
    FragU f;
    f.u[0] = *(const v4u*)(p);
    f.u[1] = *(const v4u*)(p + 16);
    return f.v;
}

__device__ inline v8f wmma_bf16(v16bf a, v16bf b, v8f c) {
    return __builtin_amdgcn_wmma_f32_16x16x32_bf16(
        /*neg_a=*/false, a, /*neg_b=*/false, b,
        /*c_mod=*/(short)0, c, /*reuse_a=*/false, /*reuse_b=*/false);
}

// ---------- problem constants ----------
#define DIN   2048
#define DOUT  2048
#define NHEAD 16
#define HD    128
#define SEQ   2048
#define BATCH 2
#define MTOT  (BATCH * SEQ)     // 4096
#define NQKV  (3 * DOUT)        // 6144

// ---------- conversion kernels ----------
__global__ void cvt_f32_bf16(const float* __restrict__ in, unsigned short* __restrict__ out, int n) {
    int i = blockIdx.x * blockDim.x + threadIdx.x;
    if (i < n) out[i] = f2bf(in[i]);
}

// in [rows][cols] f32 -> out [cols][rows] bf16 (weight transpose so K is contiguous)
__global__ void cvt_transpose_bf16(const float* __restrict__ in, unsigned short* __restrict__ out,
                                   int rows, int cols) {
    int i = blockIdx.x * blockDim.x + threadIdx.x;
    if (i < rows * cols) {
        int r = i / cols, c = i % cols;
        out[c * rows + r] = f2bf(in[i]);
    }
}

// ---------- GEMM 1: qkv = x @ Wqkv, scatter to Q/K [B,H,N,hd] and V^T [B,H,hd,N] ----------
#define LSTR 40   // 32 + 8 halves pad (80B, 16B aligned)

__global__ __launch_bounds__(256)
void gemm_qkv_kernel(const unsigned short* __restrict__ X,    // [4096][2048] bf16
                     const unsigned short* __restrict__ WT,   // [6144][2048] bf16 (W^T)
                     unsigned short* __restrict__ qd,
                     unsigned short* __restrict__ kd,
                     unsigned short* __restrict__ vd) {
    __shared__ __align__(16) unsigned short As[2][128 * LSTR];
    __shared__ __align__(16) unsigned short Bs[2][128 * LSTR];
    const int K = DIN;
    int m0 = blockIdx.y * 128;
    int n0 = blockIdx.x * 128;
    int tid = threadIdx.x, wave = tid >> 5, lane = tid & 31;
    int wm = (wave & 3) * 32, wn = (wave >> 2) * 64;
    int colb = lane & 15, rh = (lane & 16) ? 8 : 0;

    // staging coordinates for this thread (2 chunks of 8 halves per tile)
    int srow[2], scc[2];
#pragma unroll
    for (int i = 0; i < 2; ++i) { int c = tid + i * 256; srow[i] = c >> 2; scc[i] = (c & 3) * 8; }

    v8f acc[2][4];
#pragma unroll
    for (int i = 0; i < 2; ++i)
#pragma unroll
        for (int j = 0; j < 4; ++j) { v8f z = {}; acc[i][j] = z; }

    // prologue: stage k0 = 0 into buffer 0
#pragma unroll
    for (int i = 0; i < 2; ++i) {
        cp_async_b128(&X[(m0 + srow[i]) * K + scc[i]],  &As[0][srow[i] * LSTR + scc[i]]);
        cp_async_b128(&WT[(n0 + srow[i]) * K + scc[i]], &Bs[0][srow[i] * LSTR + scc[i]]);
    }
    cp_async_wait0();
    __syncthreads();

    int cur = 0;
    for (int k0 = 0; k0 < K; k0 += 32) {
        int nxt = cur ^ 1;
        if (k0 + 32 < K) {      // prefetch next K tile while we compute
#pragma unroll
            for (int i = 0; i < 2; ++i) {
                cp_async_b128(&X[(m0 + srow[i]) * K + k0 + 32 + scc[i]],
                              &As[nxt][srow[i] * LSTR + scc[i]]);
                cp_async_b128(&WT[(n0 + srow[i]) * K + k0 + 32 + scc[i]],
                              &Bs[nxt][srow[i] * LSTR + scc[i]]);
            }
        }
        v16bf af[2], bf[4];
#pragma unroll
        for (int i = 0; i < 2; ++i) af[i] = load_frag(As[cur], LSTR, wm + i * 16 + (lane & 15), 0);
#pragma unroll
        for (int j = 0; j < 4; ++j) bf[j] = load_frag(Bs[cur], LSTR, wn + j * 16 + (lane & 15), 0);
#pragma unroll
        for (int i = 0; i < 2; ++i)
#pragma unroll
            for (int j = 0; j < 4; ++j) acc[i][j] = wmma_bf16(af[i], bf[j], acc[i][j]);
        cp_async_wait0();
        __syncthreads();
        cur = nxt;
    }

#pragma unroll
    for (int i = 0; i < 2; ++i)
#pragma unroll
        for (int j = 0; j < 4; ++j)
#pragma unroll
            for (int r = 0; r < 8; ++r) {
                int mrow = m0 + wm + i * 16 + r + rh;
                int c    = n0 + wn + j * 16 + colb;
                int btk = mrow >> 11, tok = mrow & (SEQ - 1);
                unsigned short hv = f2bf(acc[i][j][r]);
                if (c < DOUT) {
                    int hh = c >> 7, hd = c & 127;
                    qd[(((btk * NHEAD + hh) * SEQ) + tok) * HD + hd] = hv;
                } else if (c < 2 * DOUT) {
                    int c2 = c - DOUT; int hh = c2 >> 7, hd = c2 & 127;
                    kd[(((btk * NHEAD + hh) * SEQ) + tok) * HD + hd] = hv;
                } else {
                    int c2 = c - 2 * DOUT; int hh = c2 >> 7, hd = c2 & 127;
                    vd[(((btk * NHEAD + hh) * HD) + hd) * SEQ + tok] = hv;   // V transposed
                }
            }
}

// ---------- fused causal flash attention ----------
#define KB 64
#define KSTR 136   // 128 + 8
#define VSTR 72    // 64 + 8
#define PSTR 72

__global__ __launch_bounds__(256)
void attn_kernel(const unsigned short* __restrict__ q_ws,   // [B,H,N,hd]
                 const unsigned short* __restrict__ k_ws,   // [B,H,N,hd]
                 const unsigned short* __restrict__ v_ws,   // [B,H,hd,N]
                 unsigned short* __restrict__ ctx) {        // [B,N,DOUT] bf16
    __shared__ __align__(16) unsigned short Ks[2][KB * KSTR];
    __shared__ __align__(16) unsigned short Vs[2][HD * VSTR];
    __shared__ __align__(16) unsigned short Ps[8][16 * PSTR];

    int bh = blockIdx.y;                 // 0..31
    int qb = blockIdx.x;                 // 0..15
    int btk = bh >> 4, h = bh & 15;
    const unsigned short* Q  = q_ws + (size_t)bh * SEQ * HD;
    const unsigned short* Kg = k_ws + (size_t)bh * SEQ * HD;
    const unsigned short* Vg = v_ws + (size_t)bh * HD * SEQ;

    int tid = threadIdx.x, wave = tid >> 5, lane = tid & 31;
    int colb = lane & 15, rh = (lane & 16) ? 8 : 0;
    int q0 = qb * 128;
    int qw = q0 + wave * 16;             // this wave's query row base

    // staging coordinates (4 chunks of 8 halves per thread per tile)
    int krow[4], kcc[4], vrow[4], vcc[4];
#pragma unroll
    for (int i = 0; i < 4; ++i) {
        int c = tid + i * 256;
        krow[i] = c >> 4; kcc[i] = (c & 15) * 8;   // K block [64][128]
        vrow[i] = c >> 3; vcc[i] = (c & 7) * 8;    // V^T block [128][64]
    }

    // Q fragments: 16 rows x 128 hd = 4 K-chunks, kept in registers
    v16bf qf[4];
#pragma unroll
    for (int c = 0; c < 4; ++c) qf[c] = load_frag(Q, HD, qw + (lane & 15), c * 32);

    v8f o[8];
#pragma unroll
    for (int j = 0; j < 8; ++j) { v8f z = {}; o[j] = z; }
    float mrow[8], lrow[8];
#pragma unroll
    for (int r = 0; r < 8; ++r) { mrow[r] = -1e30f; lrow[r] = 0.0f; }

    const float scale = 0.08838834764831845f;   // 1/sqrt(128)
    int kend = q0 + 128;

    // prologue: stage key block 0 into buffer 0
#pragma unroll
    for (int i = 0; i < 4; ++i) {
        cp_async_b128(&Kg[krow[i] * HD + kcc[i]],  &Ks[0][krow[i] * KSTR + kcc[i]]);
        cp_async_b128(&Vg[vrow[i] * SEQ + vcc[i]], &Vs[0][vrow[i] * VSTR + vcc[i]]);
    }
    cp_async_wait0();
    __syncthreads();

    int cur = 0;
    for (int kb = 0; kb < kend; kb += KB) {
        int nxt = cur ^ 1;
        if (kb + KB < kend) {   // prefetch next key block while computing
#pragma unroll
            for (int i = 0; i < 4; ++i) {
                cp_async_b128(&Kg[(kb + KB + krow[i]) * HD + kcc[i]],
                              &Ks[nxt][krow[i] * KSTR + kcc[i]]);
                cp_async_b128(&Vg[vrow[i] * SEQ + kb + KB + vcc[i]],
                              &Vs[nxt][vrow[i] * VSTR + vcc[i]]);
            }
        }

        // S = Q K^T  (4 key tiles x 4 hd chunks)
        v8f s[4];
#pragma unroll
        for (int t = 0; t < 4; ++t) {
            v8f z = {}; s[t] = z;
#pragma unroll
            for (int c = 0; c < 4; ++c) {
                v16bf kf = load_frag(Ks[cur], KSTR, t * 16 + (lane & 15), c * 32);
                s[t] = wmma_bf16(qf[c], kf, s[t]);
            }
        }

        bool diag = (kb + KB > qw);
#pragma unroll
        for (int t = 0; t < 4; ++t)
#pragma unroll
            for (int r = 0; r < 8; ++r) {
                float v = s[t][r] * scale;
                if (diag) {
                    int key = kb + t * 16 + colb;
                    int qi  = qw + r + rh;
                    if (key > qi) v = -1e30f;
                }
                s[t][r] = v;
            }

        // online softmax per row; write P (bf16) to per-wave LDS scratch
#pragma unroll
        for (int r = 0; r < 8; ++r) {
            float mx = s[0][r];
#pragma unroll
            for (int t = 1; t < 4; ++t) mx = fmaxf(mx, s[t][r]);
#pragma unroll
            for (int d = 1; d < 16; d <<= 1) mx = fmaxf(mx, __shfl_xor(mx, d, 32));
            float mnew  = fmaxf(mrow[r], mx);
            float alpha = __expf(mrow[r] - mnew);
            float sum = 0.0f;
#pragma unroll
            for (int t = 0; t < 4; ++t) {
                float p = __expf(s[t][r] - mnew);
                s[t][r] = p;
                sum += p;
            }
#pragma unroll
            for (int d = 1; d < 16; d <<= 1) sum += __shfl_xor(sum, d, 32);
            lrow[r] = lrow[r] * alpha + sum;
            mrow[r] = mnew;
#pragma unroll
            for (int j = 0; j < 8; ++j) o[j][r] *= alpha;
#pragma unroll
            for (int t = 0; t < 4; ++t)
                Ps[wave][(r + rh) * PSTR + t * 16 + colb] = f2bf(s[t][r]);
        }

        // O += P @ V   (8 hd tiles x 2 key chunks)
        v16bf pf[2];
        pf[0] = load_frag(&Ps[wave][0], PSTR, (lane & 15), 0);
        pf[1] = load_frag(&Ps[wave][0], PSTR, (lane & 15), 32);
#pragma unroll
        for (int j = 0; j < 8; ++j)
#pragma unroll
            for (int c = 0; c < 2; ++c) {
                v16bf vf = load_frag(Vs[cur], VSTR, j * 16 + (lane & 15), c * 32);
                o[j] = wmma_bf16(pf[c], vf, o[j]);
            }
        cp_async_wait0();
        __syncthreads();
        cur = nxt;
    }

    // normalize and write ctx (bf16) in [B,N,DOUT] layout
#pragma unroll
    for (int j = 0; j < 8; ++j)
#pragma unroll
        for (int r = 0; r < 8; ++r) {
            float val = o[j][r] / lrow[r];
            int qi = qw + r + rh;
            int hd = j * 16 + colb;
            ctx[((size_t)(btk * SEQ + qi)) * DOUT + h * HD + hd] = f2bf(val);
        }
}

// ---------- GEMM 2: out = ctx @ Wout + b ----------
__global__ __launch_bounds__(256)
void gemm_out_kernel(const unsigned short* __restrict__ CTX,  // [4096][2048] bf16
                     const unsigned short* __restrict__ WT,   // [2048][2048] bf16 (Wout^T)
                     const float* __restrict__ bias,
                     float* __restrict__ out) {               // [4096][2048] f32
    __shared__ __align__(16) unsigned short As[2][128 * LSTR];
    __shared__ __align__(16) unsigned short Bs[2][128 * LSTR];
    const int K = DOUT;
    int m0 = blockIdx.y * 128;
    int n0 = blockIdx.x * 128;
    int tid = threadIdx.x, wave = tid >> 5, lane = tid & 31;
    int wm = (wave & 3) * 32, wn = (wave >> 2) * 64;
    int colb = lane & 15, rh = (lane & 16) ? 8 : 0;

    int srow[2], scc[2];
#pragma unroll
    for (int i = 0; i < 2; ++i) { int c = tid + i * 256; srow[i] = c >> 2; scc[i] = (c & 3) * 8; }

    v8f acc[2][4];
#pragma unroll
    for (int i = 0; i < 2; ++i)
#pragma unroll
        for (int j = 0; j < 4; ++j) { v8f z = {}; acc[i][j] = z; }

#pragma unroll
    for (int i = 0; i < 2; ++i) {
        cp_async_b128(&CTX[(m0 + srow[i]) * K + scc[i]], &As[0][srow[i] * LSTR + scc[i]]);
        cp_async_b128(&WT[(n0 + srow[i]) * K + scc[i]],  &Bs[0][srow[i] * LSTR + scc[i]]);
    }
    cp_async_wait0();
    __syncthreads();

    int cur = 0;
    for (int k0 = 0; k0 < K; k0 += 32) {
        int nxt = cur ^ 1;
        if (k0 + 32 < K) {
#pragma unroll
            for (int i = 0; i < 2; ++i) {
                cp_async_b128(&CTX[(m0 + srow[i]) * K + k0 + 32 + scc[i]],
                              &As[nxt][srow[i] * LSTR + scc[i]]);
                cp_async_b128(&WT[(n0 + srow[i]) * K + k0 + 32 + scc[i]],
                              &Bs[nxt][srow[i] * LSTR + scc[i]]);
            }
        }
        v16bf af[2], bf[4];
#pragma unroll
        for (int i = 0; i < 2; ++i) af[i] = load_frag(As[cur], LSTR, wm + i * 16 + (lane & 15), 0);
#pragma unroll
        for (int j = 0; j < 4; ++j) bf[j] = load_frag(Bs[cur], LSTR, wn + j * 16 + (lane & 15), 0);
#pragma unroll
        for (int i = 0; i < 2; ++i)
#pragma unroll
            for (int j = 0; j < 4; ++j) acc[i][j] = wmma_bf16(af[i], bf[j], acc[i][j]);
        cp_async_wait0();
        __syncthreads();
        cur = nxt;
    }

#pragma unroll
    for (int i = 0; i < 2; ++i)
#pragma unroll
        for (int j = 0; j < 4; ++j)
#pragma unroll
            for (int r = 0; r < 8; ++r) {
                int mr = m0 + wm + i * 16 + r + rh;
                int c  = n0 + wn + j * 16 + colb;
                out[(size_t)mr * DIN + c] = acc[i][j][r] + bias[c];
            }
}

// ---------- launch ----------
extern "C" void kernel_launch(void* const* d_in, const int* in_sizes, int n_in,
                              void* d_out, int out_size, void* d_ws, size_t ws_size,
                              hipStream_t stream) {
    const float* x    = (const float*)d_in[0];   // [2,2048,2048]
    const float* Wqkv = (const float*)d_in[1];   // [2048,6144]
    const float* Wout = (const float*)d_in[2];   // [2048,2048]
    const float* bout = (const float*)d_in[3];   // [2048]
    float* out = (float*)d_out;

    char* ws = (char*)d_ws;
    unsigned short* xbf   = (unsigned short*)(ws);                       // 16 MB
    unsigned short* wqkvT = (unsigned short*)(ws + 16777216ull);         // 24 MB
    unsigned short* woutT = (unsigned short*)(ws + 41943040ull);         //  8 MB
    unsigned short* q_ws  = (unsigned short*)(ws + 50331648ull);         // 16 MB
    unsigned short* k_ws  = (unsigned short*)(ws + 67108864ull);         // 16 MB
    unsigned short* v_ws  = (unsigned short*)(ws + 83886080ull);         // 16 MB
    unsigned short* ctx   = (unsigned short*)(ws + 100663296ull);        // 16 MB

    const int nX = MTOT * DIN;          // 8388608
    cvt_f32_bf16<<<(nX + 255) / 256, 256, 0, stream>>>(x, xbf, nX);
    cvt_transpose_bf16<<<(DIN * NQKV + 255) / 256, 256, 0, stream>>>(Wqkv, wqkvT, DIN, NQKV);
    cvt_transpose_bf16<<<(DOUT * DIN + 255) / 256, 256, 0, stream>>>(Wout, woutT, DOUT, DIN);

    gemm_qkv_kernel<<<dim3(NQKV / 128, MTOT / 128), 256, 0, stream>>>(xbf, wqkvT, q_ws, k_ws, v_ws);
    attn_kernel<<<dim3(SEQ / 128, BATCH * NHEAD), 256, 0, stream>>>(q_ws, k_ws, v_ws, ctx);
    gemm_out_kernel<<<dim3(DIN / 128, MTOT / 128), 256, 0, stream>>>(ctx, woutT, bout, out);
}
